// GenCRnn_33767032881950
// MI455X (gfx1250) — matrix operand order repaired and checked by reading, not compiled, not run on hardware
//
#include <hip/hip_runtime.h>

// ---------------- problem dims ----------------
#define B_  64
#define S_  2048
#define F_  128
#define H_  512
#define G_  2048          // 4*H
#define K1_ 256           // 2*F (fc1 K)
#define KL_ 1024          // LSTM K ([act | h])

#define NWG   64
#define TPB   64          // 2 waves/WG -> 128 waves total, 2 waves per WGP

typedef __attribute__((ext_vector_type(16))) __bf16 v16bf;
typedef __attribute__((ext_vector_type(8)))  __bf16 v8bf;
typedef __attribute__((ext_vector_type(8)))  float  v8f;

union FragU { v16bf v; v8bf h[2]; };

// ---------------- workspace layout (bytes) ----------------
#define OFF_WFC1 0u          // [512][256] bf16   = 262144
#define OFF_W1   262144u     // [2048][1024] bf16 = 4194304  (Wih1|Whh1)
#define OFF_W2   4456448u    // [2048][1024] bf16 = 4194304  (Wih2|Whh2)
#define OFF_WFC2 8650752u    // [128][512] bf16   = 131072
#define OFF_CAT  8781824u    // [64][256] bf16    = 32768   ([x_t | prev_gen])
#define OFF_ACT1 8814592u    // [64][512] bf16    = 65536
#define OFF_H1A  8880128u    // [64][512] bf16    = 65536
#define OFF_H1B  8945664u
#define OFF_H2A  9011200u
#define OFF_H2B  9076736u
#define OFF_BAR  9142272u    // 2 x u32 barrier state

// ---------------- LDS layout (per wave; 2 waves per WG) ----------------
#define LDS_W1       0u        // [64][1024] bf16 = 131072 (this wave's LSTM1 slice)
#define LDS_FC1      131072u   // [16][256]  bf16 = 8192
#define LDS_FC2      139264u   // [16][512]  bf16 = 16384
#define LDS_PER_WAVE 155648u
#define LDS_TOTAL    (2u * LDS_PER_WAVE)   // 311296 <= 320KB per WGP

// ---------------- helpers ----------------
__device__ __forceinline__ v16bf load_fragA(const __bf16* p) {
  // per-lane: 16 contiguous K values at +0..15 (caller applied the 0/16 half-offset)
  FragU f;
  f.h[0] = *(const v8bf*)(p);
  f.h[1] = *(const v8bf*)(p + 8);
  return f.v;
}
__device__ __forceinline__ v16bf load_fragB(const __bf16* p) {
  // per-lane: 16 contiguous K values (caller applied the 0/16 half-offset)
  return *(const v16bf*)(p);
}
__device__ __forceinline__ v8f wmma_bf16(v16bf a, v16bf b, v8f c) {
  return __builtin_amdgcn_wmma_f32_16x16x32_bf16(false, a, false, b, (short)0, c,
                                                 false, false);
}
__device__ __forceinline__ float sigm(float x) { return 1.0f / (1.0f + __expf(-x)); }
__device__ __forceinline__ float tanh_f(float x) {
  float e = __expf(-2.0f * fabsf(x));
  float t = (1.0f - e) / (1.0f + e);
  return copysignf(t, x);
}

__device__ __forceinline__ void gridbar(unsigned* bar) {
  __syncthreads();
  if (threadIdx.x == 0) {
    __threadfence();
    unsigned g = __atomic_load_n(bar + 1, __ATOMIC_RELAXED);
    unsigned arrived = __atomic_add_fetch(bar, 1u, __ATOMIC_ACQ_REL);
    if (arrived == NWG) {
      __atomic_store_n(bar, 0u, __ATOMIC_RELAXED);
      __atomic_fetch_add(bar + 1, 1u, __ATOMIC_ACQ_REL);
    } else {
      while (__atomic_load_n(bar + 1, __ATOMIC_ACQUIRE) == g)
        __builtin_amdgcn_s_sleep(2);
    }
    __threadfence();
  }
  __syncthreads();
}

// Shared LSTM GEMM + elementwise. B-operand row pointers are precomputed by the
// caller so address-space inference keeps LDS (ds_*) vs global paths distinct.
__device__ __forceinline__ void lstm_core(
    const __bf16* __restrict__ a0,   // this wave's A rows into layer input
    const __bf16* __restrict__ a1,   // this wave's A rows into previous h
    const __bf16* Bi, const __bf16* Bf, const __bf16* Bg, const __bf16* Bo,
    __bf16* __restrict__ hOut,
    v8f& cst, float bi, float bf, float bg, float bo,
    int mi, int hc, int lane, bool last,
    float* __restrict__ hF, float* __restrict__ cF)
{
  const int l15  = lane & 15;
  const int hi16 = (lane & 16) ? 1 : 0;
  v8f gi, gf, gg, go;
#pragma unroll
  for (int r = 0; r < 8; ++r) { gi[r] = bi; gf[r] = bf; gg[r] = bg; go[r] = bo; }

#pragma unroll 4
  for (int kk = 0; kk < 512; kk += 32) {          // input-projection half
    v16bf a = load_fragA(a0 + kk);
    gi = wmma_bf16(a, load_fragB(Bi + kk), gi);
    gf = wmma_bf16(a, load_fragB(Bf + kk), gf);
    gg = wmma_bf16(a, load_fragB(Bg + kk), gg);
    go = wmma_bf16(a, load_fragB(Bo + kk), go);
  }
#pragma unroll 4
  for (int kk = 0; kk < 512; kk += 32) {          // recurrent half
    v16bf a = load_fragA(a1 + kk);
    gi = wmma_bf16(a, load_fragB(Bi + 512 + kk), gi);
    gf = wmma_bf16(a, load_fragB(Bf + 512 + kk), gf);
    gg = wmma_bf16(a, load_fragB(Bg + 512 + kk), gg);
    go = wmma_bf16(a, load_fragB(Bo + 512 + kk), go);
  }

#pragma unroll
  for (int r = 0; r < 8; ++r) {
    float cv = sigm(gf[r]) * cst[r] + sigm(gi[r]) * tanh_f(gg[r]);
    float hv = sigm(go[r]) * tanh_f(cv);
    cst[r] = cv;
    int row = mi * 16 + r + hi16 * 8;
    int col = hc * 16 + l15;
    hOut[row * H_ + col] = (__bf16)hv;
    if (last) { hF[row * H_ + col] = hv; cF[row * H_ + col] = cv; }
  }
}

// ---------------- one-shot setup kernels ----------------
__global__ void convert_weights(const float* __restrict__ Wfc1f,
                                const float* __restrict__ Wih1, const float* __restrict__ Whh1,
                                const float* __restrict__ Wih2, const float* __restrict__ Whh2,
                                const float* __restrict__ Wfc2f, char* __restrict__ ws)
{
  __bf16* Wfc1 = (__bf16*)(ws + OFF_WFC1);
  __bf16* W1   = (__bf16*)(ws + OFF_W1);
  __bf16* W2   = (__bf16*)(ws + OFF_W2);
  __bf16* Wfc2 = (__bf16*)(ws + OFF_WFC2);
  const int R0 = H_ * K1_;            // 131072
  const int R1 = G_ * KL_;            // 2097152
  const int R3 = F_ * H_;             // 65536
  const int total = R0 + 2 * R1 + R3;
  for (int idx = blockIdx.x * blockDim.x + threadIdx.x; idx < total;
       idx += gridDim.x * blockDim.x) {
    if (idx < R0) {
      Wfc1[idx] = (__bf16)Wfc1f[idx];
    } else if (idx < R0 + R1) {
      int d = idx - R0; int n = d >> 10, k = d & 1023;
      W1[d] = (__bf16)((k < H_) ? Wih1[n * H_ + k] : Whh1[n * H_ + (k - H_)]);
    } else if (idx < R0 + 2 * R1) {
      int d = idx - R0 - R1; int n = d >> 10, k = d & 1023;
      W2[d] = (__bf16)((k < H_) ? Wih2[n * H_ + k] : Whh2[n * H_ + (k - H_)]);
    } else {
      int d = idx - R0 - 2 * R1;
      Wfc2[d] = (__bf16)Wfc2f[d];
    }
  }
}

__global__ void init_state(const float* __restrict__ x, const float* __restrict__ pg0,
                           const float* __restrict__ h1_0, const float* __restrict__ h2_0,
                           char* __restrict__ ws)
{
  __bf16* cat = (__bf16*)(ws + OFF_CAT);
  __bf16* h1A = (__bf16*)(ws + OFF_H1A);
  __bf16* h2A = (__bf16*)(ws + OFF_H2A);
  unsigned* bar = (unsigned*)(ws + OFF_BAR);
  int idx = blockIdx.x * blockDim.x + threadIdx.x;
  if (idx < 2) bar[idx] = 0u;
  if (idx < B_ * K1_) {
    int b = idx >> 8, c = idx & 255;
    cat[idx] = (__bf16)((c < F_) ? x[(size_t)b * (S_ * F_) + c]
                                 : pg0[b * F_ + (c - F_)]);
  }
  if (idx < B_ * H_) {
    h1A[idx] = (__bf16)h1_0[idx];
    h2A[idx] = (__bf16)h2_0[idx];
  }
}

// ---------------- persistent RNN kernel ----------------
__global__ __launch_bounds__(TPB, 1) void rnn_persistent(
    const float* __restrict__ x,     // [B][S][F]
    const float* __restrict__ c1_0, const float* __restrict__ c2_0,
    const float* __restrict__ bfc1, const float* __restrict__ b1,
    const float* __restrict__ b2, const float* __restrict__ bfc2,
    char* __restrict__ ws, float* __restrict__ out)
{
  const __bf16* Wfc1 = (const __bf16*)(ws + OFF_WFC1);
  const __bf16* W1   = (const __bf16*)(ws + OFF_W1);
  const __bf16* W2   = (const __bf16*)(ws + OFF_W2);
  const __bf16* Wfc2 = (const __bf16*)(ws + OFF_WFC2);
  __bf16* cat  = (__bf16*)(ws + OFF_CAT);
  __bf16* act1 = (__bf16*)(ws + OFF_ACT1);
  __bf16* h1buf[2] = { (__bf16*)(ws + OFF_H1A), (__bf16*)(ws + OFF_H1B) };
  __bf16* h2buf[2] = { (__bf16*)(ws + OFF_H2A), (__bf16*)(ws + OFF_H2B) };
  unsigned* bar = (unsigned*)(ws + OFF_BAR);

  const int lane  = threadIdx.x & 31;
  const int gw    = blockIdx.x * (TPB / 32) + (threadIdx.x >> 5); // 0..127
  const int l15   = lane & 15;
  const int hi16  = (lane & 16) ? 1 : 0;
  const int koffA = hi16 ? 16 : 0;
  const int koffB = hi16 ? 16 : 0;
  const int mi = gw >> 5;   // row tile 0..3
  const int hc = gw & 31;   // h-column tile 0..31
  const int mi3 = gw >> 3, n3 = gw & 7;                     // fc2 tile (gw<32)

  // ---- per-wave LDS-resident weight slices (persist across all 2048 steps)
  extern __shared__ char smem[];
  char* lbase = smem + (threadIdx.x >> 5) * LDS_PER_WAVE;
  __bf16* lW1  = (__bf16*)(lbase + LDS_W1);    // rows: gate*16 + j, stride 1024
  __bf16* lFC1 = (__bf16*)(lbase + LDS_FC1);   // rows: j, stride 256
  __bf16* lFC2 = (__bf16*)(lbase + LDS_FC2);   // rows: j, stride 512

  for (int r = 0; r < 64; ++r) {               // LSTM1 slice (i,f,g,o x 16 rows)
    int g = r >> 4, j = r & 15;
    const __bf16* src = W1 + (size_t)(g * 512 + hc * 16 + j) * KL_;
    __bf16* dst = lW1 + r * KL_;
    for (int k = lane * 8; k < KL_; k += 256)
      *(v8bf*)(dst + k) = *(const v8bf*)(src + k);
  }
  for (int j = 0; j < 16; ++j) {               // fc1 slice
    const __bf16* src = Wfc1 + (size_t)(hc * 16 + j) * K1_;
    __bf16* dst = lFC1 + j * K1_;
    for (int k = lane * 8; k < K1_; k += 256)
      *(v8bf*)(dst + k) = *(const v8bf*)(src + k);
  }
  if (gw < 32) {                               // fc2 slice
    for (int j = 0; j < 16; ++j) {
      const __bf16* src = Wfc2 + (size_t)(n3 * 16 + j) * H_;
      __bf16* dst = lFC2 + j * H_;
      for (int k = lane * 8; k < H_; k += 256)
        *(v8bf*)(dst + k) = *(const v8bf*)(src + k);
    }
  }

  // persistent register-resident cell state tiles
  v8f c1r, c2r;
#pragma unroll
  for (int r = 0; r < 8; ++r) {
    int row = mi * 16 + r + hi16 * 8;
    int col = hc * 16 + l15;
    c1r[r] = c1_0[row * H_ + col];
    c2r[r] = c2_0[row * H_ + col];
  }
  const float bfc1v = bfc1[hc * 16 + l15];
  const float b1i = b1[hc*16+l15], b1f = b1[512+hc*16+l15],
              b1g = b1[1024+hc*16+l15], b1o = b1[1536+hc*16+l15];
  const float b2i = b2[hc*16+l15], b2f = b2[512+hc*16+l15],
              b2g = b2[1024+hc*16+l15], b2o = b2[1536+hc*16+l15];
  const float bfc2v = (gw < 32) ? bfc2[n3 * 16 + l15] : 0.0f;

  float* h1F = out + (size_t)B_ * S_ * F_;
  float* c1F = h1F + B_ * H_;
  float* h2F = c1F + B_ * H_;
  float* c2F = h2F + B_ * H_;

  // LDS B-operand row pointers (LSTM1, fc1, fc2)
  const __bf16* L1i = lW1 + (size_t)(l15     ) * KL_ + koffB;
  const __bf16* L1f = lW1 + (size_t)(16 + l15) * KL_ + koffB;
  const __bf16* L1g = lW1 + (size_t)(32 + l15) * KL_ + koffB;
  const __bf16* L1o = lW1 + (size_t)(48 + l15) * KL_ + koffB;
  const __bf16* LB1 = lFC1 + (size_t)l15 * K1_ + koffB;
  const __bf16* LB2 = lFC2 + (size_t)l15 * H_ + koffB;

  // global B-operand row pointers (LSTM2, streamed from L2)
  const int nrow = hc * 16 + l15;
  const __bf16* G2i = W2 + (size_t)(nrow        ) * KL_ + koffB;
  const __bf16* G2f = W2 + (size_t)(nrow + 512  ) * KL_ + koffB;
  const __bf16* G2g = W2 + (size_t)(nrow + 1024 ) * KL_ + koffB;
  const __bf16* G2o = W2 + (size_t)(nrow + 1536 ) * KL_ + koffB;

  for (int t = 0; t < S_; ++t) {
    const bool last = (t == S_ - 1);
    __bf16* h1in  = h1buf[t & 1];
    __bf16* h1out = h1buf[(t + 1) & 1];
    __bf16* h2in  = h2buf[t & 1];
    __bf16* h2out = h2buf[(t + 1) & 1];

    // ---- phase A: act1 = relu(cat @ Wfc1^T + bfc1), B from LDS
    {
      v8f acc;
#pragma unroll
      for (int r = 0; r < 8; ++r) acc[r] = bfc1v;
      const __bf16* A = cat + (size_t)(mi * 16 + l15) * K1_ + koffA;
#pragma unroll
      for (int kk = 0; kk < K1_; kk += 32)
        acc = wmma_bf16(load_fragA(A + kk), load_fragB(LB1 + kk), acc);
#pragma unroll
      for (int r = 0; r < 8; ++r) {
        float v = acc[r] > 0.0f ? acc[r] : 0.0f;
        int row = mi * 16 + r + hi16 * 8;
        act1[row * H_ + hc * 16 + l15] = (__bf16)v;
      }
    }
    gridbar(bar);

    // ---- phase B: LSTM1, B-operand from LDS
    lstm_core(act1 + (size_t)(mi * 16 + l15) * H_ + koffA,
              h1in + (size_t)(mi * 16 + l15) * H_ + koffA,
              L1i, L1f, L1g, L1o, h1out,
              c1r, b1i, b1f, b1g, b1o, mi, hc, lane, last, h1F, c1F);
    gridbar(bar);

    // ---- phase C: LSTM2, B-operand streamed from L2
    lstm_core(h1out + (size_t)(mi * 16 + l15) * H_ + koffA,
              h2in  + (size_t)(mi * 16 + l15) * H_ + koffA,
              G2i, G2f, G2g, G2o, h2out,
              c2r, b2i, b2f, b2g, b2o, mi, hc, lane, last, h2F, c2F);
    gridbar(bar);

    // ---- phase D: fc2 on waves 0..31 (B from LDS), x staging on waves 32..63
    if (gw < 32) {
      v8f acc;
#pragma unroll
      for (int r = 0; r < 8; ++r) acc[r] = bfc2v;
      const __bf16* A = h2out + (size_t)(mi3 * 16 + l15) * H_ + koffA;
#pragma unroll 4
      for (int kk = 0; kk < H_; kk += 32)
        acc = wmma_bf16(load_fragA(A + kk), load_fragB(LB2 + kk), acc);
#pragma unroll
      for (int r = 0; r < 8; ++r) {
        int row = mi3 * 16 + r + hi16 * 8;
        int col = n3 * 16 + l15;
        float g = acc[r];
        out[(size_t)row * (S_ * F_) + (size_t)t * F_ + col] = g;  // generated
        cat[row * K1_ + F_ + col] = (__bf16)g;                    // feedback
      }
    } else if (gw < 64 && t + 1 < S_) {
      int i = (gw - 32) * 32 + lane;
#pragma unroll
      for (int j = 0; j < 8; ++j) {
        int e = i * 8 + j;
        int b = e >> 7, f = e & 127;
        cat[b * K1_ + f] =
            (__bf16)x[(size_t)b * (S_ * F_) + (size_t)(t + 1) * F_ + f];
      }
    }
    gridbar(bar);
  }
}

// ---------------- launch ----------------
extern "C" void kernel_launch(void* const* d_in, const int* in_sizes, int n_in,
                              void* d_out, int out_size, void* d_ws, size_t ws_size,
                              hipStream_t stream) {
  (void)in_sizes; (void)n_in; (void)out_size; (void)ws_size;
  const float* x     = (const float*)d_in[0];
  const float* pg0   = (const float*)d_in[1];
  const float* h1_0  = (const float*)d_in[2];
  const float* c1_0  = (const float*)d_in[3];
  const float* h2_0  = (const float*)d_in[4];
  const float* c2_0  = (const float*)d_in[5];
  const float* Wfc1f = (const float*)d_in[6];
  const float* bfc1  = (const float*)d_in[7];
  const float* Wih1  = (const float*)d_in[8];
  const float* Whh1  = (const float*)d_in[9];
  const float* b1    = (const float*)d_in[10];
  const float* Wih2  = (const float*)d_in[11];
  const float* Whh2  = (const float*)d_in[12];
  const float* b2    = (const float*)d_in[13];
  const float* Wfc2f = (const float*)d_in[14];
  const float* bfc2  = (const float*)d_in[15];
  char* ws = (char*)d_ws;
  float* out = (float*)d_out;

  // Raise the dynamic-LDS cap (304KB/WG); host attribute set, not a stream op,
  // so graph-capture safe and deterministic.
  (void)hipFuncSetAttribute((const void*)rnn_persistent,
                            hipFuncAttributeMaxDynamicSharedMemorySize,
                            (int)LDS_TOTAL);

  convert_weights<<<1024, 256, 0, stream>>>(Wfc1f, Wih1, Whh1, Wih2, Whh2, Wfc2f, ws);
  init_state<<<256, 128, 0, stream>>>(x, pg0, h1_0, h2_0, ws);
  rnn_persistent<<<NWG, TPB, LDS_TOTAL, stream>>>(x, c1_0, c2_0, bfc1, b1, b2,
                                                  bfc2, ws, out);
}
